// GroupedQueryAttention_90056874262695
// MI455X (gfx1250) — compile-verified, matrix operand order checked by reading, not run on hardware
//
#include <hip/hip_runtime.h>
#include <hip/hip_bf16.h>

typedef __attribute__((ext_vector_type(16))) __bf16 v16bf;
typedef __attribute__((ext_vector_type(8)))  __bf16 bf8;
typedef __attribute__((ext_vector_type(4)))  __bf16 bf4;
typedef __attribute__((ext_vector_type(8)))  float  v8f;
typedef __attribute__((ext_vector_type(4)))  unsigned int u32x4;
typedef __attribute__((ext_vector_type(8)))  int i32x8;
typedef __attribute__((ext_vector_type(4)))  int i32x4;

#define BATCH 2
#define SEQ   2048
#define EMB   2048
#define NH    32
#define NKV   8
#define HD    64

// ---------------- WMMA helpers ----------------
static __device__ __forceinline__ v8f wmma_bf16(v16bf a, v16bf b, v8f c) {
    return __builtin_amdgcn_wmma_f32_16x16x32_bf16(
        false, a, false, b, (short)0, c, false, false);
}
static __device__ __forceinline__ v16bf cat8(bf8 lo, bf8 hi) {
    return __builtin_shufflevector(lo, hi, 0,1,2,3,4,5,6,7,8,9,10,11,12,13,14,15);
}

// ---------------- TDM 2D tile load: global bf16 -> padded LDS ----------------
// Loads a (tileRows x 32) bf16 tile whose rows are `stride` elements apart in
// global memory into LDS with a 40-half pitch (pad 4 DWORDs every 16 DWORDs),
// per the D# layout in the CDNA5 ISA (CDNA5 ISA).
static __device__ __forceinline__ void tdm_load_tile(
    unsigned ldsOff, const __bf16* gptr, unsigned rowLen,
    unsigned tileRows, unsigned tensorRows)
{
    const unsigned long long ga = (unsigned long long)(size_t)gptr;
    u32x4 g0;
    g0.x = 1u;                                          // count=1 (valid), user mode
    g0.y = ldsOff;                                      // lds_addr (bytes)
    g0.z = (unsigned)ga;                                // global_addr[31:0]
    g0.w = (unsigned)((ga >> 32) & 0x01FFFFFFu) | (2u << 30); // addr[56:32] | type=2
    i32x8 g1;
    g1[0] = (int)((1u << 16)     // data_size = 2 bytes
                | (1u << 20)     // pad_enable
                | (3u << 22)     // pad_interval: every 16 DWORDs (one 32-half row)
                | (3u << 25));   // pad_amount: 4 DWORDs (8 halves) -> pitch 40
    g1[1] = (int)((rowLen & 0xFFFFu) << 16);                               // tensor_dim0 lo
    g1[2] = (int)(((rowLen >> 16) & 0xFFFFu) | ((tensorRows & 0xFFFFu) << 16)); // dim0 hi | dim1 lo
    g1[3] = (int)(((tensorRows >> 16) & 0xFFFFu) | (32u << 16));           // dim1 hi | tile_dim0=32
    g1[4] = (int)(tileRows & 0xFFFFu);                                     // tile_dim1 | tile_dim2=0
    g1[5] = (int)rowLen;                                                   // tensor_dim0_stride lo
    g1[6] = 0;                                                             // stride hi | dim1_stride lo
    g1[7] = 0;
    const i32x4 z4 = {0, 0, 0, 0};
#if __clang_major__ >= 23
    const i32x8 z8 = {0, 0, 0, 0, 0, 0, 0, 0};
    __builtin_amdgcn_tensor_load_to_lds(g0, g1, z4, z4, z8, 0);
#else
    __builtin_amdgcn_tensor_load_to_lds(g0, g1, z4, z4, 0);
#endif
}
static __device__ __forceinline__ unsigned lds_off(const void* p) {
    return (unsigned)(size_t)p;     // low 32 bits of an LDS address = byte offset
}

// ================= bf16 hi/lo packing kernels ================================
__global__ void pack_rows(const float* __restrict__ src,
                          __bf16* __restrict__ hi, __bf16* __restrict__ lo)
{
    const size_t i = (size_t)blockIdx.x * 256 + threadIdx.x;   // float4 index
    const float4 v = *(const float4*)(src + i * 4);
    const float f[4] = {v.x, v.y, v.z, v.w};
    bf4 h, l;
    #pragma unroll
    for (int j = 0; j < 4; ++j) {
        h[j] = (__bf16)f[j];
        l[j] = (__bf16)(f[j] - (float)h[j]);
    }
    *(bf4*)(hi + i * 4) = h;
    *(bf4*)(lo + i * 4) = l;
}

// W[K][N] fp32 -> Wt_hi/Wt_lo [N][K] bf16 (transposed so GEMM B rows are K-major)
__global__ void pack_wt(const float* __restrict__ W,
                        __bf16* __restrict__ th, __bf16* __restrict__ tl,
                        int K, int N)
{
    const size_t idx = (size_t)blockIdx.x * 256 + threadIdx.x; // over K*N, N fastest
    const int    n   = (int)(idx % N);
    const size_t k   = idx / N;
    const float  f   = W[idx];
    const __bf16 h   = (__bf16)f;
    th[(size_t)n * K + k] = h;
    tl[(size_t)n * K + k] = (__bf16)(f - (float)h);
}

// ================= GEMM: C[M,N] = A[M,K] @ B^T[N,K], bf16 hi/lo x3 ==========
// A pre-packed [M][K] hi/lo bf16; B pre-packed transposed [N][K] hi/lo bf16.
// Double-buffered LDS tiles filled by the Tensor Data Mover (wave 0 issues,
// TENSORcnt + workgroup barrier synchronize), WMMA consumes.
#define BM 128
#define BN 64
#define BK 32
#define APITCH 40   // 32 halves + 8 pad = 80B rows (16B multiple)

__global__ __launch_bounds__(256)
void gemm_tdm_bf16x3(const __bf16* __restrict__ Ahg, const __bf16* __restrict__ Alg,
                     const __bf16* __restrict__ Bhg, const __bf16* __restrict__ Blg,
                     float* __restrict__ C, int M, int N, int K)
{
    __shared__ __bf16 Ah[2][BM][APITCH];
    __shared__ __bf16 Al[2][BM][APITCH];
    __shared__ __bf16 Bh[2][BN][APITCH];
    __shared__ __bf16 Bl[2][BN][APITCH];

    const int t    = threadIdx.x;
    const int lane = t & 31;
    const int w    = t >> 5;
    const int wm   = w & 3;          // 4 waves along M
    const int wn   = w >> 2;         // 2 waves along N
    const int hf   = lane >> 4;
    const int l16  = lane & 15;
    const int m0   = blockIdx.y * BM;
    const int n0   = blockIdx.x * BN;

    v8f acc[2][2] = {};

    const int nsteps = K / BK;
    if (w == 0) {   // wave-level DMA issue (EXEC-independent)
        tdm_load_tile(lds_off(&Ah[0][0][0]), Ahg + (size_t)m0 * K, K, BM, M);
        tdm_load_tile(lds_off(&Al[0][0][0]), Alg + (size_t)m0 * K, K, BM, M);
        tdm_load_tile(lds_off(&Bh[0][0][0]), Bhg + (size_t)n0 * K, K, BN, N);
        tdm_load_tile(lds_off(&Bl[0][0][0]), Blg + (size_t)n0 * K, K, BN, N);
    }

    for (int i = 0; i < nsteps; ++i) {
        const int buf = i & 1;
        if (w == 0) {
            if (i + 1 < nsteps) {   // prefetch next k-tile into the other buffer
                const int k1 = (i + 1) * BK;
                tdm_load_tile(lds_off(&Ah[buf ^ 1][0][0]), Ahg + (size_t)m0 * K + k1, K, BM, M);
                tdm_load_tile(lds_off(&Al[buf ^ 1][0][0]), Alg + (size_t)m0 * K + k1, K, BM, M);
                tdm_load_tile(lds_off(&Bh[buf ^ 1][0][0]), Bhg + (size_t)n0 * K + k1, K, BN, N);
                tdm_load_tile(lds_off(&Bl[buf ^ 1][0][0]), Blg + (size_t)n0 * K + k1, K, BN, N);
                __builtin_amdgcn_s_wait_tensorcnt(4);   // current tile's 4 DMAs done
            } else {
                __builtin_amdgcn_s_wait_tensorcnt(0);
            }
        }
        __syncthreads();            // TDM results visible to all waves

        v16bf ahi[2], alo[2], bhi[2], blo[2];
        #pragma unroll
        for (int mi = 0; mi < 2; ++mi) {
            const int row = wm * 32 + mi * 16 + l16;
            ahi[mi] = cat8(*(const bf8*)&Ah[buf][row][hf * 8],
                           *(const bf8*)&Ah[buf][row][hf * 8 + 16]);
            alo[mi] = cat8(*(const bf8*)&Al[buf][row][hf * 8],
                           *(const bf8*)&Al[buf][row][hf * 8 + 16]);
        }
        #pragma unroll
        for (int ni = 0; ni < 2; ++ni) {
            const int col = wn * 32 + ni * 16 + l16;
            bhi[ni] = cat8(*(const bf8*)&Bh[buf][col][hf * 8],
                           *(const bf8*)&Bh[buf][col][hf * 8 + 16]);
            blo[ni] = cat8(*(const bf8*)&Bl[buf][col][hf * 8],
                           *(const bf8*)&Bl[buf][col][hf * 8 + 16]);
        }
        #pragma unroll
        for (int mi = 0; mi < 2; ++mi)
            #pragma unroll
            for (int ni = 0; ni < 2; ++ni) {
                acc[mi][ni] = wmma_bf16(ahi[mi], bhi[ni], acc[mi][ni]);
                acc[mi][ni] = wmma_bf16(ahi[mi], blo[ni], acc[mi][ni]);
                acc[mi][ni] = wmma_bf16(alo[mi], bhi[ni], acc[mi][ni]);
            }
        __syncthreads();            // done with buf before TDM overwrites it
    }

    #pragma unroll
    for (int mi = 0; mi < 2; ++mi)
        #pragma unroll
        for (int ni = 0; ni < 2; ++ni) {
            const int colg = n0 + wn * 32 + ni * 16 + l16;
            #pragma unroll
            for (int r = 0; r < 8; ++r) {
                const int rowg = m0 + wm * 32 + mi * 16 + r + hf * 8;
                C[(size_t)rowg * N + colg] = acc[mi][ni][r];
            }
        }
}

// ================= RoPE =====================================================
__global__ void rope_q_kernel(float* __restrict__ q,
                              const float* __restrict__ cosT,
                              const float* __restrict__ sinT)
{
    int idx = blockIdx.x * 256 + threadIdx.x;       // over B*NH*S*(HD/2)
    const int d = idx & 31;  idx >>= 5;
    const int s = idx & (SEQ - 1); idx >>= 11;
    const int h = idx & (NH - 1);  idx >>= 5;
    const int b = idx;
    float* p = q + ((size_t)(b * SEQ + s)) * (NH * HD) + h * HD;
    const float x1 = p[d], x2 = p[d + 32];
    const float c1 = cosT[s * HD + d],      s1 = sinT[s * HD + d];
    const float c2 = cosT[s * HD + d + 32], s2 = sinT[s * HD + d + 32];
    p[d]      = x1 * c1 - x2 * s1;
    p[d + 32] = x2 * c2 + x1 * s2;
}

__global__ void rope_kv_kernel(const float* __restrict__ kin,
                               const float* __restrict__ vin,
                               float* __restrict__ kout,
                               float* __restrict__ vout,
                               const float* __restrict__ cosT,
                               const float* __restrict__ sinT)
{
    int idx = blockIdx.x * 256 + threadIdx.x;       // over B*NKV*S*(HD/2)
    const int d = idx & 31;  idx >>= 5;
    const int s = idx & (SEQ - 1); idx >>= 11;
    const int h = idx & (NKV - 1); idx >>= 3;
    const int b = idx;
    const size_t src = (size_t)(b * SEQ + s) * (NKV * HD) + h * HD;
    const size_t dst = ((size_t)(b * NKV + h) * SEQ + s) * HD;
    const float x1 = kin[src + d], x2 = kin[src + d + 32];
    const float c1 = cosT[s * HD + d],      s1 = sinT[s * HD + d];
    const float c2 = cosT[s * HD + d + 32], s2 = sinT[s * HD + d + 32];
    kout[dst + d]      = x1 * c1 - x2 * s1;
    kout[dst + d + 32] = x2 * c2 + x1 * s2;
    vout[dst + d]      = vin[src + d];
    vout[dst + d + 32] = vin[src + d + 32];
}

// ================= Flash attention (causal, GQA) =============================
__global__ __launch_bounds__(256)
void flash_attn(const float* __restrict__ Q,     // [B][S][NH*HD] (rope'd)
                const float* __restrict__ Kn,    // [B][NKV][S][HD]
                const float* __restrict__ Vn,    // [B][NKV][S][HD]
                float* __restrict__ O)           // [B][S][NH*HD]
{
    __shared__ __bf16 Kt[32][72];        // [key][d]
    __shared__ __bf16 Vt[HD][40];        // [d][key]
    __shared__ __bf16 Pl[8][16][40];     // per-wave P scratch

    const int t = threadIdx.x, lane = t & 31, w = t >> 5;
    const int hf = lane >> 4, l16 = lane & 15;
    const int h  = blockIdx.y, b = blockIdx.z;
    const int hk = h >> 2;                       // GROUP = 4
    const int q0 = blockIdx.x * 128 + w * 16;

    const float* qrow = Q + ((size_t)b * SEQ + (q0 + l16)) * (NH * HD) + h * HD;
    v16bf aq[2];
    #pragma unroll
    for (int f = 0; f < 2; ++f)
        #pragma unroll
        for (int hh = 0; hh < 16; ++hh) {
            const int d = f * 32 + ((hh >> 3) << 4) + hf * 8 + (hh & 7);
            aq[f][hh] = (__bf16)qrow[d];
        }

    v8f   o[4] = {};
    float mr[8], lr[8];
    #pragma unroll
    for (int r = 0; r < 8; ++r) { mr[r] = -3e38f; lr[r] = 0.f; }

    const size_t kvbase = ((size_t)b * NKV + hk) * SEQ * HD;
    const int ktEnd = blockIdx.x * 4 + 4;

    for (int kt = 0; kt < ktEnd; ++kt) {
        __syncthreads();
        for (int i = t; i < 32 * HD; i += 256) {
            const int key = i >> 6, d = i & 63;
            const size_t g = kvbase + (size_t)(kt * 32 + key) * HD + d;
            Kt[key][d] = (__bf16)Kn[g];
            Vt[d][key] = (__bf16)Vn[g];
            if (kt + 1 < ktEnd) {
                __builtin_prefetch(&Kn[g + 32 * HD], 0, 1);
                __builtin_prefetch(&Vn[g + 32 * HD], 0, 1);
            }
        }
        __syncthreads();
        if (kt * 32 > q0 + 15) continue;

        v8f s[2] = {};
        #pragma unroll
        for (int ks = 0; ks < 2; ++ks)
            #pragma unroll
            for (int f = 0; f < 2; ++f) {
                const int key = ks * 16 + l16;
                v16bf bk = cat8(*(const bf8*)&Kt[key][f * 32 + hf * 8],
                                *(const bf8*)&Kt[key][f * 32 + hf * 8 + 16]);
                s[ks] = wmma_bf16(aq[f], bk, s[ks]);
            }

        float ps0[8], ps1[8], alpha[8];
        #pragma unroll
        for (int r = 0; r < 8; ++r) {
            const int qg = q0 + r + hf * 8;
            float v0 = s[0][r] * 0.125f;
            float v1 = s[1][r] * 0.125f;
            if (kt * 32 + l16 > qg)      v0 = -3e38f;
            if (kt * 32 + 16 + l16 > qg) v1 = -3e38f;
            float mx = fmaxf(v0, v1);
            mx = fmaxf(mx, __shfl_xor(mx, 1));
            mx = fmaxf(mx, __shfl_xor(mx, 2));
            mx = fmaxf(mx, __shfl_xor(mx, 4));
            mx = fmaxf(mx, __shfl_xor(mx, 8));
            const float mnew = fmaxf(mr[r], mx);
            const float a    = __expf(mr[r] - mnew);
            const float p0   = __expf(v0 - mnew);
            const float p1   = __expf(v1 - mnew);
            float rs = p0 + p1;
            rs += __shfl_xor(rs, 1);
            rs += __shfl_xor(rs, 2);
            rs += __shfl_xor(rs, 4);
            rs += __shfl_xor(rs, 8);
            lr[r] = lr[r] * a + rs;
            mr[r] = mnew;
            alpha[r] = a;
            ps0[r] = p0; ps1[r] = p1;
        }
        #pragma unroll
        for (int f = 0; f < 4; ++f)
            #pragma unroll
            for (int r = 0; r < 8; ++r) o[f][r] *= alpha[r];

        #pragma unroll
        for (int r = 0; r < 8; ++r) {
            const int m = r + hf * 8;
            Pl[w][m][l16]      = (__bf16)ps0[r];
            Pl[w][m][16 + l16] = (__bf16)ps1[r];
        }
        const v16bf ap = cat8(*(const bf8*)&Pl[w][l16][hf * 8],
                              *(const bf8*)&Pl[w][l16][hf * 8 + 16]);

        #pragma unroll
        for (int f = 0; f < 4; ++f) {
            v16bf bv = cat8(*(const bf8*)&Vt[f * 16 + l16][hf * 8],
                            *(const bf8*)&Vt[f * 16 + l16][hf * 8 + 16]);
            o[f] = wmma_bf16(ap, bv, o[f]);
        }
    }

    float* obase = O + (size_t)b * SEQ * (NH * HD);
    #pragma unroll
    for (int r = 0; r < 8; ++r) {
        const float inv = 1.f / lr[r];
        const int srow = q0 + r + hf * 8;
        #pragma unroll
        for (int f = 0; f < 4; ++f)
            obase[(size_t)srow * (NH * HD) + h * HD + f * 16 + l16] = o[f][r] * inv;
    }
}

// ================= driver ====================================================
extern "C" void kernel_launch(void* const* d_in, const int* in_sizes, int n_in,
                              void* d_out, int out_size, void* d_ws, size_t ws_size,
                              hipStream_t stream)
{
    const float* x    = (const float*)d_in[0];
    const float* cosT = (const float*)d_in[2];
    const float* sinT = (const float*)d_in[3];
    const float* Wq   = (const float*)d_in[4];
    const float* Wk   = (const float*)d_in[5];
    const float* Wv   = (const float*)d_in[6];
    const float* Wo   = (const float*)d_in[7];

    float* out   = (float*)d_out;                              // [B,S,E]
    float* knext = out + (size_t)BATCH * SEQ * EMB;            // [B,NKV,S,HD]
    float* vnext = knext + (size_t)BATCH * NKV * SEQ * HD;

    // workspace layout (256B aligned slabs)
    char* wp = (char*)d_ws;
    auto alloc = [&](size_t bytes) {
        void* p = (void*)wp;
        wp += (bytes + 255) & ~(size_t)255;
        return p;
    };
    const size_t MS = (size_t)BATCH * SEQ;          // 4096 rows
    float*  qws = (float*)alloc(MS * EMB * 4);
    float*  kws = (float*)alloc(MS * NKV * HD * 4);
    float*  vws = (float*)alloc(MS * NKV * HD * 4);
    float*  ows = (float*)alloc(MS * EMB * 4);
    __bf16* xh  = (__bf16*)alloc(MS * EMB * 2);
    __bf16* xl  = (__bf16*)alloc(MS * EMB * 2);
    __bf16* oh  = (__bf16*)alloc(MS * EMB * 2);
    __bf16* ol  = (__bf16*)alloc(MS * EMB * 2);
    __bf16* wqh = (__bf16*)alloc((size_t)EMB * EMB * 2);
    __bf16* wql = (__bf16*)alloc((size_t)EMB * EMB * 2);
    __bf16* wkh = (__bf16*)alloc((size_t)EMB * NKV * HD * 2);
    __bf16* wkl = (__bf16*)alloc((size_t)EMB * NKV * HD * 2);
    __bf16* wvh = (__bf16*)alloc((size_t)EMB * NKV * HD * 2);
    __bf16* wvl = (__bf16*)alloc((size_t)EMB * NKV * HD * 2);
    __bf16* woh = (__bf16*)alloc((size_t)EMB * EMB * 2);
    __bf16* wol = (__bf16*)alloc((size_t)EMB * EMB * 2);

    const int M = (int)MS;
    dim3 blk(256);

    // ---- pre-pack activations and (transposed) weights to bf16 hi/lo ----
    pack_rows<<<(unsigned)(MS * EMB / 4 / 256), blk, 0, stream>>>(x, xh, xl);
    pack_wt<<<(unsigned)((size_t)EMB * EMB / 256), blk, 0, stream>>>(Wq, wqh, wql, EMB, EMB);
    pack_wt<<<(unsigned)((size_t)EMB * NKV * HD / 256), blk, 0, stream>>>(Wk, wkh, wkl, EMB, NKV * HD);
    pack_wt<<<(unsigned)((size_t)EMB * NKV * HD / 256), blk, 0, stream>>>(Wv, wvh, wvl, EMB, NKV * HD);
    pack_wt<<<(unsigned)((size_t)EMB * EMB / 256), blk, 0, stream>>>(Wo, woh, wol, EMB, EMB);

    // ---- projections (TDM-fed, double-buffered bf16x3 WMMA GEMMs) ----
    gemm_tdm_bf16x3<<<dim3(EMB / BN, M / BM), blk, 0, stream>>>(xh, xl, wqh, wql, qws, M, EMB, EMB);
    gemm_tdm_bf16x3<<<dim3((NKV * HD) / BN, M / BM), blk, 0, stream>>>(xh, xl, wkh, wkl, kws, M, NKV * HD, EMB);
    gemm_tdm_bf16x3<<<dim3((NKV * HD) / BN, M / BM), blk, 0, stream>>>(xh, xl, wvh, wvl, vws, M, NKV * HD, EMB);

    // ---- RoPE (+ emit next_k / next_v directly into d_out) ----
    rope_q_kernel<<<(BATCH * NH * SEQ * 32) / 256, blk, 0, stream>>>(qws, cosT, sinT);
    rope_kv_kernel<<<(BATCH * NKV * SEQ * 32) / 256, blk, 0, stream>>>(kws, vws, knext, vnext, cosT, sinT);

    // ---- causal GQA flash attention ----
    flash_attn<<<dim3(SEQ / 128, NH, BATCH), blk, 0, stream>>>(qws, knext, vnext, ows);

    // ---- output projection ----
    pack_rows<<<(unsigned)(MS * EMB / 4 / 256), blk, 0, stream>>>(ows, oh, ol);
    gemm_tdm_bf16x3<<<dim3(EMB / BN, M / BM), blk, 0, stream>>>(oh, ol, woh, wol, out, M, EMB, EMB);
}
